// Model_22892175688160
// MI455X (gfx1250) — compile-verified, matrix operand order checked by reading, not run on hardware
//
#include <hip/hip_runtime.h>
#include <hip/hip_bf16.h>
#include <math.h>

// ---------------------------------------------------------------------------
// Model: B=16, T=1024, D=128, H=4, DH=32.  All matmuls via v_wmma_f32_16x16x32_f16.
// ---------------------------------------------------------------------------
#define Bsz 16
#define Tsz 1024
#define Dsz 128
#define Hh  4
#define DH  32
#define MROWS (Bsz * Tsz)      // 16384

typedef __attribute__((ext_vector_type(16))) _Float16 v16h;
typedef __attribute__((ext_vector_type(8)))  _Float16 v8h;
typedef __attribute__((ext_vector_type(8)))  float    v8f;

// ---------------------------------------------------------------------------
// WMMA helpers.  A-operand layout (16x32 f16, ISA 7.12.2):
//   lane l: row M = l&15; K slots = {koff..koff+7} U {koff+16..koff+23}, koff = (l>=16)*8
// B-operand uses the same register layout applied to B^T (so a row-major W[N,K]
// weight matrix — i.e. x @ W.T — loads with the identical tile loader).
// C/D layout: lane l holds column N=l&15; rows M = i (+8 for upper half) in acc[i].
// ---------------------------------------------------------------------------
__device__ __forceinline__ v8f wmma_f16(v16h a, v16h b, v8f c) {
  return __builtin_amdgcn_wmma_f32_16x16x32_f16(false, a, false, b, (short)0, c,
                                                false, false);
}

// 16x32 operand tile from a row-major matrix (global), leading dim ld (f16 elems)
__device__ __forceinline__ v16h load_a16x32(const _Float16* __restrict__ base, int ld) {
  int lane = threadIdx.x & 31;
  int row  = lane & 15;
  int koff = (lane >> 4) << 3;           // 0 or 8
  const _Float16* p = base + row * ld + koff;
  v8h lo = *(const v8h*)(p);
  v8h hi = *(const v8h*)(p + 16);
  v16h r;
#pragma unroll
  for (int e = 0; e < 8; ++e) { r[e] = lo[e]; r[8 + e] = hi[e]; }
  return r;
}

// Same but for a wave-private 16x32 LDS tile (ld fixed = 32) -> ds_load path
__device__ __forceinline__ v16h load_a16x32_lds(const _Float16* p0) {
  int lane = threadIdx.x & 31;
  int row  = lane & 15;
  int koff = (lane >> 4) << 3;
  const _Float16* p = p0 + row * 32 + koff;
  v16h r;
#pragma unroll
  for (int e = 0; e < 8; ++e) { r[e] = p[e]; r[8 + e] = p[16 + e]; }
  return r;
}

// Transposed operand tile: logical Mt[r][c] = base[c*ld + r]  (for V^T in P.V)
__device__ __forceinline__ v16h load_aT16x32(const _Float16* __restrict__ base, int ld) {
  int lane = threadIdx.x & 31;
  int row  = lane & 15;
  int koff = (lane >> 4) << 3;
  v16h r;
#pragma unroll
  for (int e = 0; e < 8; ++e) {
    r[e]     = base[(koff + e) * ld + row];
    r[8 + e] = base[(koff + 16 + e) * ld + row];
  }
  return r;
}

// ---------------------------------------------------------------------------
// K0: f32 -> f16 conversion (weights)
// ---------------------------------------------------------------------------
__global__ void cvt_f16_kernel(const float* __restrict__ in, _Float16* __restrict__ out, int n) {
  int i = blockIdx.x * 256 + threadIdx.x;
  if (i < n) out[i] = (_Float16)in[i];
}

// ---------------------------------------------------------------------------
// K1: per-batch run-length encode.  One block per batch.
// ---------------------------------------------------------------------------
__global__ __launch_bounds__(256)
void rle_kernel(const float* __restrict__ x, const float* __restrict__ dc_w,
                const float* __restrict__ dc_b,
                int* __restrict__ transid, int* __restrict__ lenid,
                int* __restrict__ posid, float* __restrict__ validm,
                float* __restrict__ dcp) {
  __shared__ int   d[Tsz];
  __shared__ int   cum[Tsz];
  __shared__ short cl[Tsz];
  __shared__ short pc[Tsz];
  __shared__ int   nb_s, total_s;
  __shared__ float inc_s;
  int b = blockIdx.x;
  const float* xb = x + b * Tsz;
  for (int t = threadIdx.x; t < Tsz; t += 256) d[t] = xb[t] > 0.f ? 1 : 0;
  __syncthreads();
  for (int t = threadIdx.x; t < Tsz; t += 256)
    transid[b * Tsz + t] = (t == 0) ? 0 : d[t - 1] * 2 + d[t];
  if (threadIdx.x == 0) {
    int nb = 0, bstart = 0, cnt1 = 0, cs = 0, prev = d[0];
    for (int t = 0; t < Tsz; ++t) {
      cnt1 += d[t];
      if (t > 0 && d[t] != prev) {
        int len = t - bstart;
        int c = len < 31 ? len : 31;
        cs += c;
        cl[nb] = (short)c;
        int p = bstart >> 4;               // bstart*64/1024
        pc[nb] = (short)(p > 63 ? 63 : p);
        cum[nb] = cs;
        ++nb;
        bstart = t;
      }
      prev = d[t];
    }
    {
      int len = Tsz - bstart;
      int c = len < 31 ? len : 31;
      cs += c;
      cl[nb] = (short)c;
      int p = bstart >> 4;
      pc[nb] = (short)(p > 63 ? 63 : p);
      cum[nb] = cs;
      ++nb;
    }
    nb_s = nb; total_s = cs; inc_s = (float)cnt1 / (float)Tsz;
  }
  __syncthreads();
  int nb = nb_s, total = total_s;
  for (int t = threadIdx.x; t < Tsz; t += 256) {
    int lo = 0, hi = nb;                   // first j with cum[j] > t
    while (lo < hi) { int mid = (lo + hi) >> 1; if (cum[mid] <= t) lo = mid + 1; else hi = mid; }
    int k = lo < nb ? lo : nb - 1;
    lenid[b * Tsz + t]  = (int)cl[k];
    posid[b * Tsz + t]  = (int)pc[k];
    validm[b * Tsz + t] = (t < total) ? 1.f : 0.f;
  }
  if (threadIdx.x < Dsz) {
    float inc = inc_s, outv = 1.f - inc;
    int j = threadIdx.x;
    // dc = [inc, out, inc, out];  dcp = dc @ dc_w.T + dc_b,  dc_w is (D,4) row-major
    dcp[b * Dsz + j] = inc * (dc_w[j * 4 + 0] + dc_w[j * 4 + 2]) +
                       outv * (dc_w[j * 4 + 1] + dc_w[j * 4 + 3]) + dc_b[j];
  }
}

// ---------------------------------------------------------------------------
// K2: embedding gathers -> f16 features. mode 0: trans_tab[trans];
// mode 1: 0.5*(blen_tab[len]+bpos_tab[pos])*valid
// ---------------------------------------------------------------------------
__global__ void build_feat_kernel(const float* __restrict__ tab0, const float* __restrict__ tab1,
                                  const int* __restrict__ id0, const int* __restrict__ id1,
                                  const float* __restrict__ validm,
                                  _Float16* __restrict__ out, int mode) {
  int i = blockIdx.x * 256 + threadIdx.x;   // over MROWS*D
  int bt = i >> 7, c = i & 127;
  float v;
  if (mode == 0) {
    v = tab0[id0[bt] * Dsz + c];
  } else {
    v = 0.5f * (tab0[id0[bt] * Dsz + c] + tab1[id1[bt] * Dsz + c]) * validm[bt];
  }
  out[i] = (_Float16)v;
}

// ---------------------------------------------------------------------------
// K3: GEMM  C[M,N] = A[M,K] (f16, ld=K) . W[N,K]^T (f16, row-major) + bias
// One wave computes a 32x64 tile: 2 A tiles x 4 B tiles, 8 accumulators.
// All 6 operand tiles of a K-step are live at once -> the compiler must use
// distinct registers, so the 12 global_load_b128 clause together with ONE
// s_wait_loadcnt, followed by 8 back-to-back v_wmma.
// block = (32,4) waves, grid = (N/64, M/128).
// ---------------------------------------------------------------------------
__global__ __launch_bounds__(128)
void gemm_bias_kernel(const _Float16* __restrict__ A, const _Float16* __restrict__ W,
                      const float* __restrict__ bias,
                      _Float16* __restrict__ out16, float* __restrict__ out32,
                      int K, int ldc) {
  int wid   = threadIdx.y;
  int lane  = threadIdx.x;
  int nbase = blockIdx.x * 64;
  int mbase = (blockIdx.y * 4 + wid) * 32;
  v8f acc[8] = {v8f{}, v8f{}, v8f{}, v8f{}, v8f{}, v8f{}, v8f{}, v8f{}};
  const _Float16* Ap = A + (size_t)mbase * K;
  for (int k0 = 0; k0 < K; k0 += 32) {
    v16h a0 = load_a16x32(Ap + k0, K);
    v16h a1 = load_a16x32(Ap + (size_t)16 * K + k0, K);
    v16h b[4];
#pragma unroll
    for (int j = 0; j < 4; ++j)
      b[j] = load_a16x32(W + (size_t)(nbase + j * 16) * K + k0, K);
    if (k0 + 32 < K) {
      __builtin_prefetch(W + (size_t)nbase * K + k0 + 32, 0, 1);
      __builtin_prefetch(Ap + k0 + 32, 0, 1);
    }
#pragma unroll
    for (int j = 0; j < 4; ++j) {
      acc[j]     = wmma_f16(a0, b[j], acc[j]);
      acc[4 + j] = wmma_f16(a1, b[j], acc[4 + j]);
    }
  }
  int colr  = lane & 15;
  int rbase = (lane >> 4) * 8;
#pragma unroll
  for (int j = 0; j < 4; ++j) {
    int col  = nbase + j * 16 + colr;
    float bv = bias[col];
#pragma unroll
    for (int i = 0; i < 8; ++i) {
      float v0 = acc[j][i] + bv;
      float v1 = acc[4 + j][i] + bv;
      size_t i0 = (size_t)(mbase + rbase + i) * ldc + col;
      size_t i1 = (size_t)(mbase + 16 + rbase + i) * ldc + col;
      if (out16) { out16[i0] = (_Float16)v0; out16[i1] = (_Float16)v1; }
      else       { out32[i0] = v0;           out32[i1] = v1; }
    }
  }
}

// ---------------------------------------------------------------------------
// K4: flash attention.  One wave per (b, h, 16-row q tile).  DH=32 == WMMA K.
// qkv layout: (B*T, 3*D) f16, cols [0,128)=Q heads, [128,256)=K, [256,384)=V.
// ---------------------------------------------------------------------------
__global__ __launch_bounds__(128)
void attn_kernel(const _Float16* __restrict__ qkv, _Float16* __restrict__ out) {
  __shared__ _Float16 pbuf[4][16 * 32];
  int wid  = threadIdx.y;
  int lane = threadIdx.x;
  int idx  = blockIdx.x * 4 + wid;         // 0 .. B*H*(T/16)-1
  int qt   = idx & 63;
  int h    = (idx >> 6) & (Hh - 1);
  int b    = idx >> 8;
  const _Float16* Qb = qkv + (size_t)(b * Tsz) * 384 + h * DH;
  const _Float16* Kb = Qb + Dsz;
  const _Float16* Vb = Qb + 2 * Dsz;
  int qbase = qt * 16;
  v16h qa = load_a16x32(Qb + (size_t)qbase * 384, 384);
  v8f o0 = {}, o1 = {};
  float m[8], l[8];
#pragma unroll
  for (int i = 0; i < 8; ++i) { m[i] = -1e30f; l[i] = 0.f; }
  _Float16* pb = pbuf[wid];
  int colr  = lane & 15;
  int rhalf = (lane >> 4) * 8;
  const float scale = 0.17677669529663687f;   // 1/sqrt(32)
  v8f z = {};
  for (int kv = 0; kv < Tsz; kv += 32) {
    v16h kb0 = load_a16x32(Kb + (size_t)kv * 384, 384);
    v16h kb1 = load_a16x32(Kb + (size_t)(kv + 16) * 384, 384);
    v8f s0 = wmma_f16(qa, kb0, z);
    v8f s1 = wmma_f16(qa, kb1, z);
    // online softmax: a q-row lives across the 16 lanes of a half-wave
#pragma unroll
    for (int i = 0; i < 8; ++i) {
      s0[i] *= scale;
      s1[i] *= scale;
      float mx = fmaxf(s0[i], s1[i]);
#pragma unroll
      for (int off = 1; off < 16; off <<= 1)
        mx = fmaxf(mx, __shfl_xor(mx, off, 32));
      float mnew = fmaxf(m[i], mx);
      float sc   = __expf(m[i] - mnew);
      float p0   = __expf(s0[i] - mnew);
      float p1   = __expf(s1[i] - mnew);
      float rsum = p0 + p1;
#pragma unroll
      for (int off = 1; off < 16; off <<= 1)
        rsum += __shfl_xor(rsum, off, 32);
      l[i] = l[i] * sc + rsum;
      m[i] = mnew;
      o0[i] *= sc; o1[i] *= sc;
      s0[i] = p0;  s1[i] = p1;
    }
    // C-layout -> A-layout for P via wave-private LDS round trip
#pragma unroll
    for (int i = 0; i < 8; ++i) {
      pb[(rhalf + i) * 32 + colr]      = (_Float16)s0[i];
      pb[(rhalf + i) * 32 + 16 + colr] = (_Float16)s1[i];
    }
    asm volatile("s_wait_dscnt 0" ::: "memory");
    v16h pa  = load_a16x32_lds(pb);
    v16h vb0 = load_aT16x32(Vb + (size_t)kv * 384, 384);       // dh cols 0..15
    v16h vb1 = load_aT16x32(Vb + (size_t)kv * 384 + 16, 384);  // dh cols 16..31
    o0 = wmma_f16(pa, vb0, o0);
    o1 = wmma_f16(pa, vb1, o1);
    asm volatile("s_wait_dscnt 0" ::: "memory");
  }
#pragma unroll
  for (int i = 0; i < 8; ++i) {
    float inv = 1.f / l[i];
    size_t row = (size_t)(b * Tsz + qbase + rhalf + i);
    out[row * Dsz + h * DH + colr]      = (_Float16)(o0[i] * inv);
    out[row * Dsz + h * DH + 16 + colr] = (_Float16)(o1[i] * inv);
  }
}

// ---------------------------------------------------------------------------
// K5: fill concat cols [256,384) with per-batch dcp
// ---------------------------------------------------------------------------
__global__ void dcp_fill_kernel(const float* __restrict__ dcp, _Float16* __restrict__ h) {
  int i = blockIdx.x * 256 + threadIdx.x;   // over MROWS*D
  int bt = i >> 7, c = i & 127;
  int b = bt >> 10;
  h[(size_t)bt * 384 + 256 + c] = (_Float16)dcp[b * Dsz + c];
}

// ---------------------------------------------------------------------------
// K6: row LayerNorm + exact GELU.  One wave per row.
// ---------------------------------------------------------------------------
__global__ __launch_bounds__(128)
void ln_gelu_kernel(const float* __restrict__ in, const float* __restrict__ g,
                    const float* __restrict__ be,
                    _Float16* __restrict__ out16, float* __restrict__ out32, int W) {
  int wid = threadIdx.y, lane = threadIdx.x;
  size_t row = (size_t)blockIdx.x * 4 + wid;
  const float* r = in + row * W;
  int n = W >> 5;                      // elems per lane (4 or 8)
  float vals[8];
  float s = 0.f, ss = 0.f;
  for (int e = 0; e < n; ++e) {
    float v = r[lane + e * 32];
    vals[e] = v; s += v; ss += v * v;
  }
#pragma unroll
  for (int off = 1; off < 32; off <<= 1) {
    s  += __shfl_xor(s, off, 32);
    ss += __shfl_xor(ss, off, 32);
  }
  float mean = s / (float)W;
  float var  = ss / (float)W - mean * mean;
  float rstd = rsqrtf(var + 1e-5f);
  for (int e = 0; e < n; ++e) {
    int c = lane + e * 32;
    float y  = (vals[e] - mean) * rstd * g[c] + be[c];
    float ge = 0.5f * y * (1.f + erff(y * 0.70710678118654752f));
    if (out16) out16[row * W + c] = (_Float16)ge;
    else       out32[row * W + c] = ge;
  }
}

// ---------------------------------------------------------------------------
// Host launcher
// ---------------------------------------------------------------------------
extern "C" void kernel_launch(void* const* d_in, const int* in_sizes, int n_in,
                              void* d_out, int out_size, void* d_ws, size_t ws_size,
                              hipStream_t stream) {
  const float* x         = (const float*)d_in[0];
  const float* trans_tab = (const float*)d_in[1];
  const float* blen_tab  = (const float*)d_in[2];
  const float* bpos_tab  = (const float*)d_in[3];
  const float* dc_w      = (const float*)d_in[4];
  const float* dc_b      = (const float*)d_in[5];
  const float* w_in1  = (const float*)d_in[6];   const float* b_in1  = (const float*)d_in[7];
  const float* w_out1 = (const float*)d_in[8];   const float* b_out1 = (const float*)d_in[9];
  const float* w_in2  = (const float*)d_in[10];  const float* b_in2  = (const float*)d_in[11];
  const float* w_out2 = (const float*)d_in[12];  const float* b_out2 = (const float*)d_in[13];
  const float* w_f1   = (const float*)d_in[14];  const float* b_f1   = (const float*)d_in[15];
  const float* g1     = (const float*)d_in[16];  const float* be1    = (const float*)d_in[17];
  const float* w_f2   = (const float*)d_in[18];  const float* b_f2   = (const float*)d_in[19];
  const float* g2     = (const float*)d_in[20];  const float* be2    = (const float*)d_in[21];

  char* wp = (char*)d_ws;
  auto take = [&](size_t bytes) -> void* {
    void* p = (void*)wp;
    wp += (bytes + 255) & ~(size_t)255;
    return p;
  };
  _Float16* hw_in1  = (_Float16*)take((size_t)384 * 128 * 2);
  _Float16* hw_out1 = (_Float16*)take((size_t)128 * 128 * 2);
  _Float16* hw_in2  = (_Float16*)take((size_t)384 * 128 * 2);
  _Float16* hw_out2 = (_Float16*)take((size_t)128 * 128 * 2);
  _Float16* hw_f1   = (_Float16*)take((size_t)256 * 384 * 2);
  _Float16* hw_f2   = (_Float16*)take((size_t)128 * 256 * 2);
  int*   transid = (int*)take((size_t)MROWS * 4);
  int*   lenid   = (int*)take((size_t)MROWS * 4);
  int*   posid   = (int*)take((size_t)MROWS * 4);
  float* validm  = (float*)take((size_t)MROWS * 4);
  float* dcp     = (float*)take((size_t)Bsz * Dsz * 4);
  _Float16* feat  = (_Float16*)take((size_t)MROWS * Dsz * 2);
  _Float16* qkv   = (_Float16*)take((size_t)MROWS * 3 * Dsz * 2);
  _Float16* attnb = (_Float16*)take((size_t)MROWS * Dsz * 2);
  _Float16* hbuf  = (_Float16*)take((size_t)MROWS * 3 * Dsz * 2);
  float*    tmp32 = (float*)take((size_t)MROWS * 256 * 4);
  _Float16* h1    = (_Float16*)take((size_t)MROWS * 256 * 2);

  dim3 wblk(32, 4);

  // weight conversion
  cvt_f16_kernel<<<(384 * 128 + 255) / 256, 256, 0, stream>>>(w_in1,  hw_in1,  384 * 128);
  cvt_f16_kernel<<<(128 * 128 + 255) / 256, 256, 0, stream>>>(w_out1, hw_out1, 128 * 128);
  cvt_f16_kernel<<<(384 * 128 + 255) / 256, 256, 0, stream>>>(w_in2,  hw_in2,  384 * 128);
  cvt_f16_kernel<<<(128 * 128 + 255) / 256, 256, 0, stream>>>(w_out2, hw_out2, 128 * 128);
  cvt_f16_kernel<<<(256 * 384 + 255) / 256, 256, 0, stream>>>(w_f1,   hw_f1,   256 * 384);
  cvt_f16_kernel<<<(128 * 256 + 255) / 256, 256, 0, stream>>>(w_f2,   hw_f2,   128 * 256);

  // RLE + dcp
  rle_kernel<<<Bsz, 256, 0, stream>>>(x, dc_w, dc_b, transid, lenid, posid, validm, dcp);

  // ---- MHA 1 (transition features) ----
  build_feat_kernel<<<MROWS * Dsz / 256, 256, 0, stream>>>(
      trans_tab, nullptr, transid, nullptr, nullptr, feat, 0);
  gemm_bias_kernel<<<dim3(384 / 64, MROWS / 128), wblk, 0, stream>>>(
      feat, hw_in1, b_in1, qkv, nullptr, 128, 384);
  attn_kernel<<<Bsz * Hh * (Tsz / 16) / 4, wblk, 0, stream>>>(qkv, attnb);
  gemm_bias_kernel<<<dim3(128 / 64, MROWS / 128), wblk, 0, stream>>>(
      attnb, hw_out1, b_out1, hbuf /* cols 0..127 */, nullptr, 128, 384);

  // ---- MHA 2 (burst features) ----
  build_feat_kernel<<<MROWS * Dsz / 256, 256, 0, stream>>>(
      blen_tab, bpos_tab, lenid, posid, validm, feat, 1);
  gemm_bias_kernel<<<dim3(384 / 64, MROWS / 128), wblk, 0, stream>>>(
      feat, hw_in2, b_in2, qkv, nullptr, 128, 384);
  attn_kernel<<<Bsz * Hh * (Tsz / 16) / 4, wblk, 0, stream>>>(qkv, attnb);
  gemm_bias_kernel<<<dim3(128 / 64, MROWS / 128), wblk, 0, stream>>>(
      attnb, hw_out2, b_out2, hbuf + 128 /* cols 128..255 */, nullptr, 128, 384);

  // dcp broadcast into cols 256..383
  dcp_fill_kernel<<<MROWS * Dsz / 256, 256, 0, stream>>>(dcp, hbuf);

  // ---- FFN ----
  gemm_bias_kernel<<<dim3(256 / 64, MROWS / 128), wblk, 0, stream>>>(
      hbuf, hw_f1, b_f1, nullptr, tmp32, 384, 256);
  ln_gelu_kernel<<<MROWS / 4, wblk, 0, stream>>>(tmp32, g1, be1, h1, nullptr, 256);
  gemm_bias_kernel<<<dim3(128 / 64, MROWS / 128), wblk, 0, stream>>>(
      h1, hw_f2, b_f2, nullptr, tmp32, 256, 128);
  ln_gelu_kernel<<<MROWS / 4, wblk, 0, stream>>>(tmp32, g2, be2, nullptr, (float*)d_out, 128);
}
